// CrossAttentionLayer_two_level_54735063220673
// MI455X (gfx1250) — matrix operand order repaired and checked by reading, not compiled, not run on hardware
//
#include <hip/hip_runtime.h>
#include <hip/hip_bf16.h>

typedef __attribute__((ext_vector_type(16))) _Float16 v16h;
typedef __attribute__((ext_vector_type(8)))  _Float16 v8h;
typedef __attribute__((ext_vector_type(8)))  float    v8f;

#define DIMS 128
#define NTOK 48
#define LDT  136   // f16 LDS row stride (pad 8 halves -> 272B, bank-spread)
#define LDF  132   // f32 LDS row stride

// ---- WMMA fragment helpers (CDNA5 v_wmma_f32_16x16x32_f16 layouts) ----

// A matrix 16x32 f16: lanes 0-15 = rows M, K in {kb..kb+15}, kb = 16*(lane>>4)
__device__ __forceinline__ v16h frag_a_lds(const _Float16* row, int k0, int lane) {
  const _Float16* p = row + k0 + ((lane >> 4) << 4);
  v8h lo = *(const v8h*)p;
  v8h hi = *(const v8h*)(p + 8);
  return __builtin_shufflevector(lo, hi, 0,1,2,3,4,5,6,7,8,9,10,11,12,13,14,15);
}

// B matrix 32x16 f16 (B = W^T, W row-major [out][in]): lane = column N,
// lanes 0-15 hold K=k0..k0+15, lanes 16-31 hold K=k0+16..k0+31 (contiguous 32B)
__device__ __forceinline__ v16h frag_b_glb(const _Float16* __restrict__ W, int col, int k0, int lane) {
  return *(const v16h*)(W + col * DIMS + k0 + ((lane >> 4) << 4));
}

__device__ __forceinline__ v8f wmma_f16(v16h a, v16h b, v8f c) {
  return __builtin_amdgcn_wmma_f32_16x16x32_f16(false, a, false, b, (short)0, c, false, false);
}

// ---- prep kernels -----------------------------------------------------

__global__ void cvt_f16_kernel(const float* __restrict__ src, _Float16* __restrict__ dst, int n) {
  int i = blockIdx.x * blockDim.x + threadIdx.x;
  if (i < n) dst[i] = (_Float16)src[i];
}

// q0s = 0.25 * (LN(query) @ Wq^T + bq)   (d^-0.5 = 1/4 folded in)
__global__ __launch_bounds__(128) void qprep_kernel(
    const float* __restrict__ query, const float* __restrict__ g1,
    const float* __restrict__ b1, const float* __restrict__ Wq,
    const float* __restrict__ bq, float* __restrict__ q0s) {
  __shared__ float qn[8][DIMS];
  int tid = threadIdx.x;
  int row = tid >> 4, g = tid & 15;
  float vals[8], s = 0.f, s2 = 0.f;
  #pragma unroll
  for (int j = 0; j < 8; ++j) {
    vals[j] = query[row * DIMS + g + 16 * j];
    s += vals[j]; s2 += vals[j] * vals[j];
  }
  #pragma unroll
  for (int off = 1; off < 16; off <<= 1) {
    s  += __shfl_xor(s,  off, 16);
    s2 += __shfl_xor(s2, off, 16);
  }
  float mean = s * (1.f / 128.f);
  float var  = s2 * (1.f / 128.f) - mean * mean;
  float rstd = rsqrtf(var + 1e-5f);
  #pragma unroll
  for (int j = 0; j < 8; ++j) {
    int c = g + 16 * j;
    qn[row][c] = (vals[j] - mean) * rstd * g1[c] + b1[c];
  }
  __syncthreads();
  for (int j = 0; j < 8; ++j) {
    int c = g + 16 * j;
    float acc = bq[c];
    for (int k = 0; k < DIMS; ++k) acc += qn[row][k] * Wq[c * DIMS + k];
    q0s[row * DIMS + c] = 0.25f * acc;
  }
}

// ---- fused main kernel: one workgroup (4 wave32) per batch tile -------

__global__ __launch_bounds__(128) void fused_main(
    const float* __restrict__ query, const float* __restrict__ tgt,
    const float* __restrict__ bk, const float* __restrict__ bv,
    const float* __restrict__ bp, const float* __restrict__ g2,
    const float* __restrict__ b2, const float* __restrict__ bf1,
    const float* __restrict__ bf2, const float* __restrict__ q0s,
    const _Float16* __restrict__ wk16, const _Float16* __restrict__ wv16,
    const _Float16* __restrict__ wp16, const _Float16* __restrict__ wf116,
    const _Float16* __restrict__ wf216, float* __restrict__ out, int Btot) {

  __shared__ __align__(16) _Float16 sT[NTOK][LDT];  // tgt tile (f16)
  __shared__ float    sK[NTOK][LDF];                // K projection (f32)
  __shared__ float    sV[NTOK][LDF];                // V projection (f32)
  __shared__ float    sX[8][DIMS];                  // post-proj residual x
  __shared__ __align__(16) _Float16 sA[16][LDT];    // 16-row f16 A staging

  const int tid  = threadIdx.x;
  const int lane = tid & 31;
  const int w    = tid >> 5;
  const int ln15 = lane & 15;

  // zero pad rows 8..15 of sA once (rows >=8 feed only discarded D rows)
  for (int i = tid; i < 8 * LDT; i += 128) (&sA[8][0])[i] = (_Float16)0.f;

  for (int b = blockIdx.x; b < Btot; b += gridDim.x) {
    // ---- stage tgt[b] (48x128 f32) into LDS as f16 ----
    const float4* src = (const float4*)(tgt + (size_t)b * (NTOK * DIMS));
    #pragma unroll
    for (int i = 0; i < 12; ++i) {
      int e = tid + i * 128;                 // 1536 float4 total
      float4 f = src[e];
      int flat = e * 4, r = flat >> 7, c = flat & 127;
      sT[r][c]     = (_Float16)f.x;
      sT[r][c + 1] = (_Float16)f.y;
      sT[r][c + 2] = (_Float16)f.z;
      sT[r][c + 3] = (_Float16)f.w;
    }
    int nb = b + gridDim.x;
    if (nb < Btot)
      __builtin_prefetch((const void*)(tgt + (size_t)nb * (NTOK * DIMS) + tid * 48), 0, 1);
    __syncthreads();

    // ---- K and V projections: [48,128]f16 @ W^T via WMMA, f32 accum ----
    for (int sel = 0; sel < 2; ++sel) {
      const _Float16* W   = sel ? wv16 : wk16;
      const float*    bia = sel ? bv   : bk;
      float (*dst)[LDF]   = sel ? sV   : sK;
      for (int t = w; t < 24; t += 4) {        // 24 tiles, 6 per wave
        int mt = t >> 3, nt = t & 7;
        v8f acc = {};
        #pragma unroll
        for (int kk = 0; kk < 4; ++kk) {
          v16h a  = frag_a_lds(&sT[mt * 16 + ln15][0], kk * 32, lane);
          v16h bb = frag_b_glb(W, nt * 16 + ln15, kk * 32, lane);
          acc = wmma_f16(a, bb, acc);
        }
        int n = nt * 16 + ln15;
        float bv_ = bia[n];
        int m0 = mt * 16 + ((lane >> 4) << 3);
        #pragma unroll
        for (int r = 0; r < 8; ++r) dst[m0 + r][n] = acc[r] + bv_;
      }
    }
    __syncthreads();

    // ---- attention: 64 (query,head) pairs x 2 threads (split v-dims) ----
    {
      int pair = tid >> 1, half = tid & 1;
      int qi = pair >> 3, h = pair & 7;        // wave0: qi<2 (global); waves1-3: windows
      float qv[16];
      #pragma unroll
      for (int d = 0; d < 16; ++d) qv[d] = q0s[qi * DIMS + h * 16 + d];
      int ntok, h0 = 0, w0 = 0;
      if (qi < 2) ntok = NTOK;
      else { int idx = qi - 2; h0 = (idx / 3) * 4; w0 = (idx % 3) * 2; ntok = 8; }

      float mx = -3.0e38f;
      for (int j = 0; j < ntok; ++j) {
        int tok = (qi < 2) ? j : ((h0 + (j >> 1)) * 6 + w0 + (j & 1));
        float s = 0.f;
        #pragma unroll
        for (int d = 0; d < 16; ++d) s += qv[d] * sK[tok][h * 16 + d];
        mx = fmaxf(mx, s);
      }
      float sum = 0.f, acc[8] = {0, 0, 0, 0, 0, 0, 0, 0};
      for (int j = 0; j < ntok; ++j) {
        int tok = (qi < 2) ? j : ((h0 + (j >> 1)) * 6 + w0 + (j & 1));
        float s = 0.f;
        #pragma unroll
        for (int d = 0; d < 16; ++d) s += qv[d] * sK[tok][h * 16 + d];
        float p = expf(s - mx);
        sum += p;
        #pragma unroll
        for (int d = 0; d < 8; ++d) acc[d] += p * sV[tok][h * 16 + half * 8 + d];
      }
      float inv = 1.f / sum;
      #pragma unroll
      for (int d = 0; d < 8; ++d)
        sA[qi][h * 16 + half * 8 + d] = (_Float16)(acc[d] * inv);
    }
    __syncthreads();

    // ---- x = query + attn @ Wp^T + bp  (8x128 via padded 16-row WMMA) ----
    #pragma unroll
    for (int tt = 0; tt < 2; ++tt) {
      int nt = w * 2 + tt;
      v8f acc = {};
      #pragma unroll
      for (int kk = 0; kk < 4; ++kk) {
        v16h a  = frag_a_lds(&sA[ln15][0], kk * 32, lane);
        v16h bb = frag_b_glb(wp16, nt * 16 + ln15, kk * 32, lane);
        acc = wmma_f16(a, bb, acc);
      }
      int n = nt * 16 + ln15;
      if (lane < 16) {                         // D rows 0..7 live in lanes 0-15
        float bia = bp[n];
        #pragma unroll
        for (int r = 0; r < 8; ++r)
          sX[r][n] = query[r * DIMS + n] + bia + acc[r];
      }
    }
    __syncthreads();

    // ---- LayerNorm(x) -> sA (f16), rows 0..7 ----
    {
      int row = tid >> 4, g = tid & 15;
      float vals[8], s = 0.f, s2 = 0.f;
      #pragma unroll
      for (int j = 0; j < 8; ++j) {
        vals[j] = sX[row][g + 16 * j];
        s += vals[j]; s2 += vals[j] * vals[j];
      }
      #pragma unroll
      for (int off = 1; off < 16; off <<= 1) {
        s  += __shfl_xor(s,  off, 16);
        s2 += __shfl_xor(s2, off, 16);
      }
      float mean = s * (1.f / 128.f);
      float var  = s2 * (1.f / 128.f) - mean * mean;
      float rstd = rsqrtf(var + 1e-5f);
      #pragma unroll
      for (int j = 0; j < 8; ++j) {
        int c = g + 16 * j;
        sA[row][c] = (_Float16)((vals[j] - mean) * rstd * g2[c] + b2[c]);
      }
    }
    __syncthreads();

    // ---- FFN layer 1: gelu(y @ Wf1^T + bf1) -> sA ----
    {
      v8f zero = {};
      v8f f1acc[2]; f1acc[0] = zero; f1acc[1] = zero;
      #pragma unroll
      for (int tt = 0; tt < 2; ++tt) {
        int nt = w * 2 + tt;
        #pragma unroll
        for (int kk = 0; kk < 4; ++kk) {
          v16h a  = frag_a_lds(&sA[ln15][0], kk * 32, lane);
          v16h bb = frag_b_glb(wf116, nt * 16 + ln15, kk * 32, lane);
          f1acc[tt] = wmma_f16(a, bb, f1acc[tt]);
        }
      }
      __syncthreads();                          // all reads of sA done
      #pragma unroll
      for (int tt = 0; tt < 2; ++tt) {
        int n = (w * 2 + tt) * 16 + ln15;
        if (lane < 16) {
          float bia = bf1[n];
          #pragma unroll
          for (int r = 0; r < 8; ++r) {
            float x = f1acc[tt][r] + bia;
            sA[r][n] = (_Float16)(0.5f * x * (1.f + erff(x * 0.70710678f)));
          }
        }
      }
    }
    __syncthreads();

    // ---- FFN layer 2 + residual -> global out ----
    {
      size_t ob = (size_t)b * (8 * DIMS);
      #pragma unroll
      for (int tt = 0; tt < 2; ++tt) {
        int nt = w * 2 + tt;
        v8f acc = {};
        #pragma unroll
        for (int kk = 0; kk < 4; ++kk) {
          v16h a  = frag_a_lds(&sA[ln15][0], kk * 32, lane);
          v16h bb = frag_b_glb(wf216, nt * 16 + ln15, kk * 32, lane);
          acc = wmma_f16(a, bb, acc);
        }
        int n = nt * 16 + ln15;
        if (lane < 16) {
          float bia = bf2[n];
          #pragma unroll
          for (int r = 0; r < 8; ++r)
            out[ob + r * DIMS + n] = sX[r][n] + acc[r] + bia;
        }
      }
    }
    // barriers at the head of the next iteration protect LDS reuse
  }
}

// ---- launcher ---------------------------------------------------------

extern "C" void kernel_launch(void* const* d_in, const int* in_sizes, int n_in,
                              void* d_out, int out_size, void* d_ws, size_t ws_size,
                              hipStream_t stream) {
  const float* query = (const float*)d_in[0];
  const float* tgt   = (const float*)d_in[1];
  const float* g1    = (const float*)d_in[2];
  const float* b1    = (const float*)d_in[3];
  const float* Wq    = (const float*)d_in[4];
  const float* bq    = (const float*)d_in[5];
  const float* Wk    = (const float*)d_in[6];
  const float* bk    = (const float*)d_in[7];
  const float* Wv    = (const float*)d_in[8];
  const float* bv    = (const float*)d_in[9];
  const float* Wp    = (const float*)d_in[10];
  const float* bp    = (const float*)d_in[11];
  const float* g2    = (const float*)d_in[12];
  const float* b2    = (const float*)d_in[13];
  const float* Wf1   = (const float*)d_in[14];
  const float* bf1   = (const float*)d_in[15];
  const float* Wf2   = (const float*)d_in[16];
  const float* bf2   = (const float*)d_in[17];

  int Btot = in_sizes[1] / (NTOK * DIMS);

  char* ws = (char*)d_ws;
  float*    q0s   = (float*)ws;                          // 4 KB
  _Float16* wk16  = (_Float16*)(ws + 4096);              // 32 KB each
  _Float16* wv16  = (_Float16*)(ws + 4096 + 1 * 32768);
  _Float16* wp16  = (_Float16*)(ws + 4096 + 2 * 32768);
  _Float16* wf116 = (_Float16*)(ws + 4096 + 3 * 32768);
  _Float16* wf216 = (_Float16*)(ws + 4096 + 4 * 32768);

  cvt_f16_kernel<<<64, 256, 0, stream>>>(Wk,  wk16,  DIMS * DIMS);
  cvt_f16_kernel<<<64, 256, 0, stream>>>(Wv,  wv16,  DIMS * DIMS);
  cvt_f16_kernel<<<64, 256, 0, stream>>>(Wp,  wp16,  DIMS * DIMS);
  cvt_f16_kernel<<<64, 256, 0, stream>>>(Wf1, wf116, DIMS * DIMS);
  cvt_f16_kernel<<<64, 256, 0, stream>>>(Wf2, wf216, DIMS * DIMS);
  qprep_kernel<<<1, 128, 0, stream>>>(query, g1, b1, Wq, bq, q0s);

  int grid = Btot < 4096 ? Btot : 4096;
  if (grid < 1) grid = 1;
  fused_main<<<grid, 128, 0, stream>>>(query, tgt, bk, bv, bp, g2, b2, bf1, bf2,
                                       q0s, wk16, wv16, wp16, wf116, wf216,
                                       (float*)d_out, Btot);
}
